// Model_25451976196110
// MI455X (gfx1250) — matrix operand order repaired and checked by reading, not compiled, not run on hardware
//
#include <hip/hip_runtime.h>
#include <hip/hip_bf16.h>
#include <stdint.h>

#define B_TOK 65536
#define H_DIM 1024
#define I_DIM 1024
#define E_NUM 16

typedef __attribute__((ext_vector_type(16))) __bf16 v16bf;
typedef __attribute__((ext_vector_type(8)))  float  v8f;

#if defined(__has_builtin)
#if __has_builtin(__builtin_amdgcn_global_load_async_to_lds_b128) && \
    __has_builtin(__builtin_amdgcn_s_wait_asynccnt)
#define USE_ASYNC_LDS 1
#endif
#endif

__device__ __forceinline__ unsigned short f2bf(float f) {
  unsigned int u = __float_as_uint(f);
  u += 0x7FFFu + ((u >> 16) & 1u);   // round-to-nearest-even
  return (unsigned short)(u >> 16);
}

// ---------------------------------------------------------------- bookkeeping
__global__ void zero_counts(int* counts) {
  if (threadIdx.x < E_NUM) counts[threadIdx.x] = 0;
}

__global__ void scan_offsets(const int* __restrict__ counts, int* __restrict__ offsets) {
  int run = 0;
  for (int e = 0; e < E_NUM; ++e) { offsets[e] = run; run += counts[e]; }
}

__global__ void scatter_kernel(const int* __restrict__ expert_of,
                               const int* __restrict__ rank_of,
                               const int* __restrict__ offsets,
                               int* __restrict__ perm) {
  int b = blockIdx.x * blockDim.x + threadIdx.x;
  int e = expert_of[b];
  perm[offsets[e] + rank_of[b]] = b;
}

// ------------------------------------------------- gating + x -> bf16 (fused)
__global__ __launch_bounds__(256)
void gate_kernel(const float* __restrict__ x, const float* __restrict__ Wg,
                 unsigned short* __restrict__ xb, int* __restrict__ counts,
                 int* __restrict__ expert_of, int* __restrict__ rank_of) {
  const int wave = threadIdx.x >> 5;
  const int lane = threadIdx.x & 31;
  const int b = blockIdx.x * 8 + wave;

  float acc[E_NUM];
#pragma unroll
  for (int e = 0; e < E_NUM; ++e) acc[e] = 0.f;

  const float* xrow = x + (size_t)b * H_DIM;
  unsigned short* xbrow = xb + (size_t)b * H_DIM;

  for (int h0 = 0; h0 < H_DIM; h0 += 32) {
    float xv = xrow[h0 + lane];
    xbrow[h0 + lane] = f2bf(xv);
#pragma unroll
    for (int e = 0; e < E_NUM; ++e)
      acc[e] += xv * Wg[e * H_DIM + h0 + lane];
  }
#pragma unroll
  for (int e = 0; e < E_NUM; ++e) {
#pragma unroll
    for (int off = 16; off > 0; off >>= 1)
      acc[e] += __shfl_xor(acc[e], off, 32);
  }
  if (lane == 0) {
    float best = acc[0]; int be = 0;
#pragma unroll
    for (int e = 1; e < E_NUM; ++e)
      if (acc[e] > best) { best = acc[e]; be = e; }   // strict > == first argmax
    int rank = atomicAdd(&counts[be], 1);
    expert_of[b] = be;
    rank_of[b] = rank;
  }
}

// ------------------------------------------------------------- We -> bf16
__global__ __launch_bounds__(256)
void convert_we(const float* __restrict__ We, unsigned short* __restrict__ Web) {
  size_t i = ((size_t)blockIdx.x * blockDim.x + threadIdx.x) * 8;
  float4 a = reinterpret_cast<const float4*>(We + i)[0];
  float4 c = reinterpret_cast<const float4*>(We + i)[1];
  uint4 o;
  o.x = (unsigned)f2bf(a.x) | ((unsigned)f2bf(a.y) << 16);
  o.y = (unsigned)f2bf(a.z) | ((unsigned)f2bf(a.w) << 16);
  o.z = (unsigned)f2bf(c.x) | ((unsigned)f2bf(c.y) << 16);
  o.w = (unsigned)f2bf(c.z) | ((unsigned)f2bf(c.w) << 16);
  reinterpret_cast<uint4*>(Web + i)[0] = o;
}

// ------------------------------------------------------------- grouped GEMM
// Block: 128 tokens x 128 outputs, K stepped by 64 per stage (2 WMMA K-halves).
// 8 waves = 2(M) x 4(N); each wave owns 64x32 = 4x2 accum tiles of 16x16.
// Double-buffered LDS, staged with GLOBAL_LOAD_ASYNC_TO_LDS_B128 when available.
#define KSTEP 64
#define PITCH 72   // ushorts per LDS row (144 B): 36*r mod 64 -> conflict-free

__device__ __forceinline__ void stage_async8(const unsigned short* g, unsigned short* l) {
#if USE_ASYNC_LDS
  typedef int v4i_vs __attribute__((vector_size(16)));   // matches builtin pointee
  __attribute__((address_space(1))) v4i_vs* gp =
      (__attribute__((address_space(1))) v4i_vs*)(uintptr_t)g;
  __attribute__((address_space(3))) v4i_vs* lp =
      (__attribute__((address_space(3))) v4i_vs*)(unsigned int)(uintptr_t)l;
  __builtin_amdgcn_global_load_async_to_lds_b128(gp, lp, 0, 0);
  __builtin_amdgcn_global_load_async_to_lds_b128(gp, lp, 16, 0);
  __builtin_amdgcn_global_load_async_to_lds_b128(gp, lp, 32, 0);
  __builtin_amdgcn_global_load_async_to_lds_b128(gp, lp, 48, 0);
  __builtin_amdgcn_global_load_async_to_lds_b128(gp, lp, 64, 0);
  __builtin_amdgcn_global_load_async_to_lds_b128(gp, lp, 80, 0);
  __builtin_amdgcn_global_load_async_to_lds_b128(gp, lp, 96, 0);
  __builtin_amdgcn_global_load_async_to_lds_b128(gp, lp, 112, 0);
#else
  (void)g; (void)l;
#endif
}

__device__ __forceinline__ void gload8(uint4 r[8], const unsigned short* g) {
  const uint4* p = reinterpret_cast<const uint4*>(g);
#pragma unroll
  for (int i = 0; i < 8; ++i) r[i] = p[i];
}

__device__ __forceinline__ void lstore8(unsigned short* l, const uint4 r[8]) {
  uint4* d = reinterpret_cast<uint4*>(l);
#pragma unroll
  for (int i = 0; i < 8; ++i) d[i] = r[i];
}

__device__ __forceinline__ void mma_stage(const unsigned short* __restrict__ Asb,
                                          const unsigned short* __restrict__ Bsb,
                                          int mw, int nw, int lo, int hi,
                                          v8f acc[4][2]) {
  union Frag { uint4 q[2]; v16bf v; };
#pragma unroll
  for (int kk = 0; kk < 2; ++kk) {          // two 32-wide K halves
    Frag a[4], b[2];
#pragma unroll
    for (int i = 0; i < 4; ++i) {           // A 16x32: lane lo = row, hi = K half
      const uint4* p =
          reinterpret_cast<const uint4*>(&Asb[(mw * 64 + i * 16 + lo) * PITCH + kk * 32]);
      a[i].q[0] = p[hi];                    // K = kk*32 + hi*8 ..
      a[i].q[1] = p[2 + hi];                // K = kk*32 + 16 + hi*8 ..
    }
#pragma unroll
    for (int j = 0; j < 2; ++j) {           // B 32x16: lane lo = col, hi = K half
      const uint4* p =
          reinterpret_cast<const uint4*>(&Bsb[(nw * 32 + j * 16 + lo) * PITCH + kk * 32]);
      b[j].q[0] = p[hi];                    // K = kk*32 + hi*8 ..
      b[j].q[1] = p[2 + hi];                // K = kk*32 + 16 + hi*8 ..
    }
#pragma unroll
    for (int i = 0; i < 4; ++i)
#pragma unroll
      for (int j = 0; j < 2; ++j)
        acc[i][j] = __builtin_amdgcn_wmma_f32_16x16x32_bf16(
            false, a[i].v, false, b[j].v, (short)0, acc[i][j], false, false);
  }
}

__global__ __launch_bounds__(256)
void moe_gemm(const unsigned short* __restrict__ xb,
              const unsigned short* __restrict__ Web,
              const int* __restrict__ perm,
              const int* __restrict__ counts,
              const int* __restrict__ offsets,
              float* __restrict__ out) {
  __shared__ unsigned short As[2][128 * PITCH];
  __shared__ unsigned short Bs[2][128 * PITCH];
  __shared__ int tokS[128];

  const int e   = blockIdx.z;
  const int n_e = counts[e];
  const int m0  = blockIdx.y * 128;
  if (m0 >= n_e) return;                    // uniform per block
  const int cnt  = min(128, n_e - m0);
  const int base = offsets[e];
  const int n0   = blockIdx.x * 128;

  const int tid  = threadIdx.x;
  const int wave = tid >> 5;
  const int lane = tid & 31;
  const int lo   = lane & 15;
  const int hi   = lane >> 4;
  const int mw   = wave >> 2;               // 0..1
  const int nw   = wave & 3;                // 0..3

  const unsigned short* wbase = Web + ((size_t)e * I_DIM + n0) * H_DIM;

  // per-thread staging row: threads 0..127 gather A (token rows),
  // threads 128..255 stream B (We rows). Uniform 8 x b128 per stage per thread.
  int mytok = -1;
  if (tid < 128) {
    mytok = (tid < cnt) ? perm[base + m0 + tid] : -1;
    tokS[tid] = mytok;
  }
  __syncthreads();

  const unsigned short* grow;
  unsigned short *l0, *l1;
  if (tid < 128) {
    int t = mytok < 0 ? 0 : mytok;
    grow = xb + (size_t)t * H_DIM;
    l0 = &As[0][tid * PITCH];
    l1 = &As[1][tid * PITCH];
  } else {
    int r = tid - 128;
    grow = wbase + (size_t)r * H_DIM;
    l0 = &Bs[0][r * PITCH];
    l1 = &Bs[1][r * PITCH];
  }

  v8f acc[4][2];
#pragma unroll
  for (int i = 0; i < 4; ++i)
#pragma unroll
    for (int j = 0; j < 2; ++j)
      acc[i][j] = (v8f){0.f, 0.f, 0.f, 0.f, 0.f, 0.f, 0.f, 0.f};

#if USE_ASYNC_LDS
  // ---- async double-buffered pipeline (LDS written by DMA path, ASYNCcnt) ----
  stage_async8(grow, l0);
  int buf = 0;
  for (int k0 = KSTEP; k0 < H_DIM; k0 += KSTEP) {
    stage_async8(grow + k0, buf ? l0 : l1);       // fill other buffer
    __builtin_amdgcn_s_wait_asynccnt(8);          // current buffer's 8 ops retired
    __syncthreads();
    mma_stage(As[buf], Bs[buf], mw, nw, lo, hi, acc);
    __syncthreads();                              // readers done before restaging
    buf ^= 1;
  }
  __builtin_amdgcn_s_wait_asynccnt(0);
  __syncthreads();
  mma_stage(As[buf], Bs[buf], mw, nw, lo, hi, acc);
#else
  // ---- software-pipelined fallback: next tile in VGPRs while computing ----
  uint4 r[8];
  gload8(r, grow);
  lstore8(l0, r);
  int buf = 0;
  for (int k0 = KSTEP; k0 < H_DIM; k0 += KSTEP) {
    gload8(r, grow + k0);                         // in flight during compute
    __syncthreads();
    mma_stage(As[buf], Bs[buf], mw, nw, lo, hi, acc);
    lstore8(buf ? l0 : l1, r);
    buf ^= 1;
  }
  __syncthreads();
  mma_stage(As[buf], Bs[buf], mw, nw, lo, hi, acc);
#endif

  // C layout: VGPR r -> M = hi*8 + r, N = lo
#pragma unroll
  for (int i = 0; i < 4; ++i) {
#pragma unroll
    for (int rr = 0; rr < 8; ++rr) {
      const int row = mw * 64 + i * 16 + hi * 8 + rr;
      int t = tokS[row];
      if (t >= 0) {
        float* op = out + (size_t)t * I_DIM + n0 + nw * 32 + lo;
#pragma unroll
        for (int j = 0; j < 2; ++j)
          op[j * 16] = acc[i][j][rr];
      }
    }
  }
}

// ---------------------------------------------------------------- launcher
extern "C" void kernel_launch(void* const* d_in, const int* in_sizes, int n_in,
                              void* d_out, int out_size, void* d_ws, size_t ws_size,
                              hipStream_t stream) {
  const float* x  = (const float*)d_in[0];   // [B, H]
  const float* Wg = (const float*)d_in[1];   // [E, H]
  const float* We = (const float*)d_in[2];   // [E, I, H]
  float* out = (float*)d_out;                // [B, I]

  char* ws = (char*)d_ws;
  unsigned short* xb  = (unsigned short*)ws; ws += (size_t)B_TOK * H_DIM * 2;
  unsigned short* Web = (unsigned short*)ws; ws += (size_t)E_NUM * I_DIM * H_DIM * 2;
  int* perm      = (int*)ws; ws += (size_t)B_TOK * 4;
  int* expert_of = (int*)ws; ws += (size_t)B_TOK * 4;
  int* rank_of   = (int*)ws; ws += (size_t)B_TOK * 4;
  int* counts    = (int*)ws; ws += (size_t)E_NUM * 4;
  int* offsets   = (int*)ws; ws += (size_t)E_NUM * 4;

  zero_counts<<<1, 32, 0, stream>>>(counts);
  gate_kernel<<<B_TOK / 8, 256, 0, stream>>>(x, Wg, xb, counts, expert_of, rank_of);
  convert_we<<<(E_NUM * I_DIM * H_DIM) / (256 * 8), 256, 0, stream>>>(We, Web);
  scan_offsets<<<1, 1, 0, stream>>>(counts, offsets);
  scatter_kernel<<<B_TOK / 256, 256, 0, stream>>>(expert_of, rank_of, offsets, perm);
  moe_gemm<<<dim3(I_DIM / 128, B_TOK / 128, E_NUM), 256, 0, stream>>>(
      xb, Web, perm, counts, offsets, out);
}
